// LSTM_24034636989034
// MI455X (gfx1250) — matrix operand order repaired and checked by reading, not compiled, not run on hardware
//
#include <hip/hip_runtime.h>
#include <hip/hip_bf16.h>

typedef _Float16 f16;
typedef __attribute__((ext_vector_type(16))) _Float16 v16h;
typedef __attribute__((ext_vector_type(8)))  float    v8f;

#define BATCH 512
#define SEQT  256
#define HID   1024
#define CLS   10

// ---------------------------------------------------------------------------
// Workspace layout (d_ws):
//   [0,               8 MiB)  : packed fp16 recurrent weights (4 gates)
//   [8 MiB,          10 MiB)  : final hidden state h_T, fp32 [512,1024]
// ---------------------------------------------------------------------------
#define WPACK_ELEMS ((size_t)4 * 1024 * 1024)          // 4 gates * 1024K * 1024N
#define HFIN_OFF    ((size_t)8 * 1024 * 1024)          // bytes

__device__ __forceinline__ float fast_sigmoid(float z) {
  return 1.0f / (1.0f + __expf(-z));
}
__device__ __forceinline__ float fast_tanh(float z) {
  float e = __expf(-2.0f * z);
  return (1.0f - e) / (1.0f + e);
}

// ---------------------------------------------------------------------------
// Pack W_{g,i,f,o}h (fp32 [1024,1024] row-major, z = h @ W) into fragment-major
// fp16 B-operand tiles for v_wmma_f32_16x16x32_f16.
// B fragment (32K x 16N), per lane 16 f16:
//   lane = (n&15) + (k&8 ? 16 : 0);  elem = (k&7) + (k&16 ? 8 : 0)
// Fragment index: (gate*64 + ntile)*32 + ktile, 512 f16 each.
// ---------------------------------------------------------------------------
__global__ void lstm_pack_weights(const float* __restrict__ Wg,
                                  const float* __restrict__ Wi,
                                  const float* __restrict__ Wf,
                                  const float* __restrict__ Wo,
                                  f16* __restrict__ wpack) {
  size_t idx = (size_t)blockIdx.x * 256 + threadIdx.x;
  if (idx >= WPACK_ELEMS) return;
  int g = (int)(idx >> 20);
  int rem = (int)(idx & ((1u << 20) - 1));
  int k = rem >> 10;
  int n = rem & 1023;
  const float* W = (g == 0) ? Wg : (g == 1) ? Wi : (g == 2) ? Wf : Wo;
  float v = W[(size_t)k * 1024 + n];

  int kt   = k >> 5;
  int kin  = k & 31;
  int nt   = n >> 4;
  int nin  = n & 15;
  int lane = nin + (((kin >> 3) & 1) ? 16 : 0);
  int e    = (kin & 7) + ((kin >= 16) ? 8 : 0);
  size_t dst = (((size_t)(g * 64 + nt) * 32 + kt) << 9) + (size_t)lane * 16 + e;
  wpack[dst] = (f16)v;
}

// ---------------------------------------------------------------------------
// Persistent per-batch-tile LSTM scan. One block owns 16 batch rows for all
// 256 timesteps; 16 waves each own 64 hidden columns (4 slices of 16).
// A-operand (h) lives in LDS in WMMA A-fragment layout, fp16.
// A fragment (16M x 32K) mapping (per CDNA5 ISA 16-bit A layout):
//   lanes 0-15 : M = lane, elems 0-7 -> K 0-7,  elems 8-15 -> K 16-23
//   lanes 16-31: M = lane-16, elems 0-7 -> K 8-15, elems 8-15 -> K 24-31
// ---------------------------------------------------------------------------
__global__ __launch_bounds__(512) void lstm_scan_kernel(
    const float* __restrict__ x,          // [512, 256]
    const f16*  __restrict__ wpack,       // packed recurrent weights
    const float* __restrict__ wxg, const float* __restrict__ wxi,
    const float* __restrict__ wxf, const float* __restrict__ wxo,
    const float* __restrict__ bg,  const float* __restrict__ bi,
    const float* __restrict__ bf_, const float* __restrict__ bo,
    float* __restrict__ h_out)            // [512, 1024] fp32
{
  __shared__ __align__(32) f16 hfrag[32 * 512];   // 32 KiB: 32 K-tiles of A-fragments
  __shared__ f16 wx_lds[4 * 1024];                // 8 KiB
  __shared__ f16 b_lds[4 * 1024];                 // 8 KiB

  const int tid  = threadIdx.x;
  const int lane = tid & 31;
  const int wave = tid >> 5;
  const int b0   = blockIdx.x * 16;

  // ---- init LDS: h = 0, stage Wx / bias as fp16 ----
  for (int i = tid; i < 32 * 512; i += 512) hfrag[i] = (f16)0.0f;
  for (int i = tid; i < 1024; i += 512) {
    wx_lds[i]        = (f16)wxg[i];
    wx_lds[1024 + i] = (f16)wxi[i];
    wx_lds[2048 + i] = (f16)wxf[i];
    wx_lds[3072 + i] = (f16)wxo[i];
    b_lds[i]         = (f16)bg[i];
    b_lds[1024 + i]  = (f16)bi[i];
    b_lds[2048 + i]  = (f16)bf_[i];
    b_lds[3072 + i]  = (f16)bo[i];
  }
  __syncthreads();

  v8f c_reg[4];
  v8f h_reg[4];
#pragma unroll
  for (int s = 0; s < 4; ++s) { c_reg[s] = (v8f){}; h_reg[s] = (v8f){}; }

  const int jbase    = wave * 64;        // this wave's h-column base
  const int ncol     = lane & 15;        // N within 16x16 D tile
  const int mrow_off = (lane >> 4) * 8;  // M offset within D tile

  for (int t = 0; t < SEQT; ++t) {
    // per-row input scalar x[b, t]
    float xv[8];
#pragma unroll
    for (int r = 0; r < 8; ++r)
      xv[r] = x[(size_t)(b0 + r + mrow_off) * SEQT + t];

    for (int s = 0; s < 4; ++s) {
      const int j0 = jbase + s * 16;
      const int nt = j0 >> 4;
      const f16* wpg = wpack + (((size_t)(0 * 64 + nt) * 32) << 9) + (size_t)lane * 16;
      const f16* wpi = wpack + (((size_t)(1 * 64 + nt) * 32) << 9) + (size_t)lane * 16;
      const f16* wpf = wpack + (((size_t)(2 * 64 + nt) * 32) << 9) + (size_t)lane * 16;
      const f16* wpo = wpack + (((size_t)(3 * 64 + nt) * 32) << 9) + (size_t)lane * 16;

      v8f ag = (v8f){}, ai = (v8f){}, af = (v8f){}, ao = (v8f){};
#pragma unroll 4
      for (int kt = 0; kt < 32; ++kt) {
        v16h a  = *(const v16h*)(&hfrag[kt * 512 + lane * 16]);
        v16h bG = *(const v16h*)(wpg + (size_t)kt * 512);
        v16h bI = *(const v16h*)(wpi + (size_t)kt * 512);
        v16h bF = *(const v16h*)(wpf + (size_t)kt * 512);
        v16h bO = *(const v16h*)(wpo + (size_t)kt * 512);
        ag = __builtin_amdgcn_wmma_f32_16x16x32_f16(false, a, false, bG, (short)0, ag, false, false);
        ai = __builtin_amdgcn_wmma_f32_16x16x32_f16(false, a, false, bI, (short)0, ai, false, false);
        af = __builtin_amdgcn_wmma_f32_16x16x32_f16(false, a, false, bF, (short)0, af, false, false);
        ao = __builtin_amdgcn_wmma_f32_16x16x32_f16(false, a, false, bO, (short)0, ao, false, false);
      }

      // gate epilogue: z = acc + x*Wx + b; c = tanh(zg)*sig(zi) + c*sig(zf);
      // h = tanh(c)*sig(zo)
      const int j = j0 + ncol;
#pragma unroll
      for (int r = 0; r < 8; ++r) {
        float zg = ag[r] + xv[r] * (float)wx_lds[j]        + (float)b_lds[j];
        float zi = ai[r] + xv[r] * (float)wx_lds[1024 + j] + (float)b_lds[1024 + j];
        float zf = af[r] + xv[r] * (float)wx_lds[2048 + j] + (float)b_lds[2048 + j];
        float zo = ao[r] + xv[r] * (float)wx_lds[3072 + j] + (float)b_lds[3072 + j];
        float gv = fast_tanh(zg);
        float iv = fast_sigmoid(zi);
        float fv = fast_sigmoid(zf);
        float ov = fast_sigmoid(zo);
        float cc = gv * iv + c_reg[s][r] * fv;
        c_reg[s][r] = cc;
        h_reg[s][r] = fast_tanh(cc) * ov;
      }
    }

    __syncthreads();  // everyone done reading old h fragments

    // scatter new h (fp16) into A-fragment layout for the next step
#pragma unroll
    for (int s = 0; s < 4; ++s) {
      const int j0 = jbase + s * 16;
#pragma unroll
      for (int r = 0; r < 8; ++r) {
        int M   = r + mrow_off;
        int j   = j0 + ncol;          // h column == next step's K index
        int kt  = j >> 5;
        int kin = j & 31;
        int half = (kin >> 3) & 1;    // K in {8..15,24..31} -> lanes 16..31
        int lp   = M + half * 16;
        int ep   = (kin < 16) ? (kin & 7) + ((kin >= 8) ? 0 : 0) : 0;
        // explicit mapping: K<8 -> e=K; 8<=K<16 -> e=K-8; 16<=K<24 -> e=K-8; K>=24 -> e=K-16
        ep = (kin < 8) ? kin : (kin < 16) ? (kin - 8) : (kin < 24) ? (kin - 8) : (kin - 16);
        hfrag[kt * 512 + lp * 16 + ep] = (f16)h_reg[s][r];
      }
    }
    __syncthreads();
  }

  // final hidden state -> global (fp32) for the projection kernel
#pragma unroll
  for (int s = 0; s < 4; ++s) {
    const int j0 = jbase + s * 16;
#pragma unroll
    for (int r = 0; r < 8; ++r) {
      int M = r + mrow_off;
      h_out[(size_t)(b0 + M) * HID + j0 + ncol] = h_reg[s][r];
    }
  }
}

// ---------------------------------------------------------------------------
// out[b, c] = h_T[b, :] @ W_ph[:, c] + b_p[c]     (512 x 10, trivial)
// ---------------------------------------------------------------------------
__global__ void lstm_proj_kernel(const float* __restrict__ h,
                                 const float* __restrict__ Wp,
                                 const float* __restrict__ bp,
                                 float* __restrict__ out) {
  int idx = blockIdx.x * 256 + threadIdx.x;
  if (idx >= BATCH * CLS) return;
  int b = idx / CLS;
  int c = idx % CLS;
  float acc = bp[c];
  const float* hb = h + (size_t)b * HID;
#pragma unroll 8
  for (int k = 0; k < HID; ++k) acc += hb[k] * Wp[(size_t)k * CLS + c];
  out[idx] = acc;
}

extern "C" void kernel_launch(void* const* d_in, const int* in_sizes, int n_in,
                              void* d_out, int out_size, void* d_ws, size_t ws_size,
                              hipStream_t stream) {
  (void)in_sizes; (void)n_in; (void)out_size; (void)ws_size;
  const float* x   = (const float*)d_in[0];
  const float* Wgx = (const float*)d_in[1];
  const float* Wgh = (const float*)d_in[2];
  const float* bg  = (const float*)d_in[3];
  const float* Wix = (const float*)d_in[4];
  const float* Wih = (const float*)d_in[5];
  const float* bi  = (const float*)d_in[6];
  const float* Wfx = (const float*)d_in[7];
  const float* Wfh = (const float*)d_in[8];
  const float* bf  = (const float*)d_in[9];
  const float* Wox = (const float*)d_in[10];
  const float* Woh = (const float*)d_in[11];
  const float* bo  = (const float*)d_in[12];
  const float* Wph = (const float*)d_in[13];
  const float* bp  = (const float*)d_in[14];

  f16*   wpack = (f16*)d_ws;
  float* hfin  = (float*)((char*)d_ws + HFIN_OFF);
  float* out   = (float*)d_out;

  // 1) pack recurrent weights to fp16 WMMA-B fragment layout (8 MiB)
  lstm_pack_weights<<<(unsigned)(WPACK_ELEMS / 256), 256, 0, stream>>>(
      Wgh, Wih, Wfh, Woh, wpack);

  // 2) persistent batch-tile LSTM scan: 32 blocks x 512 threads (16 waves)
  lstm_scan_kernel<<<BATCH / 16, 512, 0, stream>>>(
      x, wpack, Wgx, Wix, Wfx, Wox, bg, bi, bf, bo, hfin);

  // 3) classifier projection
  lstm_proj_kernel<<<(BATCH * CLS + 255) / 256, 256, 0, stream>>>(
      hfin, Wph, bp, out);
}
